// HGT_60224031424836
// MI455X (gfx1250) — matrix operand order repaired and checked by reading, not compiled, not run on hardware
//
#include <hip/hip_runtime.h>
#include <hip/hip_bf16.h>
#include <math.h>

#define HID   256
#define NHEAD 8
#define DHEAD 32

typedef __attribute__((ext_vector_type(16))) __bf16 v16bf;
typedef __attribute__((ext_vector_type(8)))  __bf16 v8bf;
typedef __attribute__((ext_vector_type(8)))  float  v8f;

__device__ __forceinline__ __bf16 f2bf(float f) {
  unsigned u = __float_as_uint(f);
  unsigned r = u + 0x7FFFu + ((u >> 16) & 1u);   // round-to-nearest-even
  unsigned short h = (unsigned short)(r >> 16);
  return __builtin_bit_cast(__bf16, h);
}

// order-preserving float->uint key for atomic max over signed floats
__device__ __forceinline__ unsigned fkey(float f) {
  unsigned u = __float_as_uint(f);
  return (u & 0x80000000u) ? ~u : (u | 0x80000000u);
}
__device__ __forceinline__ float funkey(unsigned k) {
  unsigned u = (k & 0x80000000u) ? (k ^ 0x80000000u) : ~k;
  return __uint_as_float(u);
}

__device__ __forceinline__ v16bf ldA(const __bf16* ap) {
  v8bf lo = *(const v8bf*)(ap);        // K = k0+kA    .. +7
  v8bf hi = *(const v8bf*)(ap + 16);   // K = k0+kA+16 .. +23
  return __builtin_shufflevector(lo, hi, 0, 1, 2, 3, 4, 5, 6, 7,
                                 8, 9, 10, 11, 12, 13, 14, 15);
}

// ---------------------------------------------------------------------------
// WMMA GEMM on pre-converted bf16 operands.
//   A  : [M,K] bf16 row-major (lda elements/row), batch stride sA
//   Wt : [N,K] bf16 (TRANSPOSED weights, contiguous K runs), batch stride sW
//   Cf : optional f32 out, Cb : optional bf16 out (same ldc/sC)
// block = 128 threads (4 waves); block tile = 64 rows x 64 cols.
// OOB handled by CLAMPING load indices (loads always legal; garbage lands only
// in never-stored D rows/cols). The k-loop is manually unswitched on the
// wave-uniform "all 4 tiles active" condition so the dominant N=256 GEMMs get
// a branch-free inner loop (tact3 => tact2 => tact1 by monotonicity).
// ---------------------------------------------------------------------------
__global__ void hgt_gemm_wmma(const __bf16* __restrict__ A, int lda, long long sA,
                              const __bf16* __restrict__ Wt, long long sW,
                              const float* __restrict__ bias,
                              float* __restrict__ Cf, __bf16* __restrict__ Cb,
                              int ldc, long long sC,
                              int M, int N, int K, int act)
{
  const int lane  = threadIdx.x & 31;
  const int wave  = threadIdx.x >> 5;
  const int tileM = blockIdx.x * 64 + wave * 16;
  const int tileN0 = blockIdx.y * 64;
  const long long b = blockIdx.z;
  A += b * sA; Wt += b * sW;

  const int mrow   = tileM + (lane & 15);
  const int mclamp = (mrow < M) ? mrow : (M - 1);
  const int ncol   = lane & 15;
  const int kA = (lane >> 4) * 8;    // A: lanes16-31 hold K+8 / K+24 halves
  const int kB = (lane >> 4) * 8;    // B: lanes16-31 hold K+8 / K+24 halves

  // wave-uniform tile activity (tile 0 always active by grid construction)
  const bool tact1 = (tileN0 + 16) < N;
  const bool tact2 = (tileN0 + 32) < N;
  const bool tact3 = (tileN0 + 48) < N;

  const __bf16* ap = A + (size_t)mclamp * lda + kA;
  const int n0 = tileN0 + ncol;
  const int nc0 = (n0      < N) ? n0      : (N - 1);
  const int nc1 = (n0 + 16 < N) ? n0 + 16 : (N - 1);
  const int nc2 = (n0 + 32 < N) ? n0 + 32 : (N - 1);
  const int nc3 = (n0 + 48 < N) ? n0 + 48 : (N - 1);
  const __bf16* bp0 = Wt + (size_t)nc0 * K + kB;
  const __bf16* bp1 = Wt + (size_t)nc1 * K + kB;
  const __bf16* bp2 = Wt + (size_t)nc2 * K + kB;
  const __bf16* bp3 = Wt + (size_t)nc3 * K + kB;

  v8f acc0 = {}, acc1 = {}, acc2 = {}, acc3 = {};

  if (tact3) {
    // fast path: all 4 tiles active -> branch-free steady state
    for (int k0 = 0; k0 < K; k0 += 32) {
      v16bf a  = ldA(ap + k0);
      v16bf b0 = ldA(bp0 + k0);
      v16bf b1 = ldA(bp1 + k0);
      v16bf b2 = ldA(bp2 + k0);
      v16bf b3 = ldA(bp3 + k0);
      acc0 = __builtin_amdgcn_wmma_f32_16x16x32_bf16(false, a, false, b0, (short)0, acc0, false, false);
      acc1 = __builtin_amdgcn_wmma_f32_16x16x32_bf16(false, a, false, b1, (short)0, acc1, false, false);
      acc2 = __builtin_amdgcn_wmma_f32_16x16x32_bf16(false, a, false, b2, (short)0, acc2, false, false);
      acc3 = __builtin_amdgcn_wmma_f32_16x16x32_bf16(false, a, false, b3, (short)0, acc3, false, false);
    }
  } else {
    // tail path: N=32 einsum (1 iter) and N=8 output linear only
    for (int k0 = 0; k0 < K; k0 += 32) {
      v16bf a = ldA(ap + k0);
      {
        v16bf b0 = ldA(bp0 + k0);
        acc0 = __builtin_amdgcn_wmma_f32_16x16x32_bf16(false, a, false, b0, (short)0, acc0, false, false);
      }
      if (tact1) {
        v16bf b1 = ldA(bp1 + k0);
        acc1 = __builtin_amdgcn_wmma_f32_16x16x32_bf16(false, a, false, b1, (short)0, acc1, false, false);
      }
      if (tact2) {
        v16bf b2 = ldA(bp2 + k0);
        acc2 = __builtin_amdgcn_wmma_f32_16x16x32_bf16(false, a, false, b2, (short)0, acc2, false, false);
      }
    }
  }

  const int mbase = tileM + (lane >> 4) * 8;
  #pragma unroll
  for (int t = 0; t < 4; ++t) {
    if (t == 1 && !tact1) continue;
    if (t == 2 && !tact2) continue;
    if (t == 3 && !tact3) continue;
    const int n = tileN0 + t * 16 + ncol;
    if (n >= N) continue;
    const float bv = bias ? bias[n] : 0.0f;
    const v8f& acc = (t == 0) ? acc0 : (t == 1) ? acc1 : (t == 2) ? acc2 : acc3;
    #pragma unroll
    for (int r = 0; r < 8; ++r) {
      const int m = mbase + r;
      if (m < M) {
        float v = acc[r] + bv;
        if (act == 1) v = fmaxf(v, 0.0f);
        const size_t off = (size_t)(b * sC) + (size_t)m * ldc + n;
        if (Cf) Cf[off] = v;
        if (Cb) Cb[off] = f2bf(v);
      }
    }
  }
}

// ---------------------------------------------------------------------------
// Conversion kernels
// ---------------------------------------------------------------------------
// f32 [K,N] row-major -> bf16 [N,K] transposed; contiguous batches of K*N
__global__ void hgt_cvt_wT(const float* __restrict__ W, __bf16* __restrict__ O,
                           int K, int N)
{
  const long long base = (long long)blockIdx.z * K * N;
  const int i = blockIdx.x * blockDim.x + threadIdx.x;
  if (i >= K * N) return;
  const int n = i % N, k = i / N;                       // coalesced f32 reads
  O[base + (size_t)n * K + k] = f2bf(W[base + (size_t)k * N + n]);
}

__global__ void hgt_cvt_bf(const float* __restrict__ x, __bf16* __restrict__ y,
                           long long n)
{
  long long i = (long long)blockIdx.x * blockDim.x + threadIdx.x;
  if (i < n) y[i] = f2bf(x[i]);
}

// ---------------------------------------------------------------------------
// Elementwise / edge kernels
// ---------------------------------------------------------------------------
__global__ void hgt_fill_f32(float* p, float v, long long n) {
  long long i = (long long)blockIdx.x * blockDim.x + threadIdx.x;
  if (i < n) p[i] = v;
}
__global__ void hgt_fill_u32(unsigned* p, unsigned v, long long n) {
  long long i = (long long)blockIdx.x * blockDim.x + threadIdx.x;
  if (i < n) p[i] = v;
}

// alpha[e,h] = (q[dst]·krel[src]) * prel[h] / sqrt(D); segment max via atomics
__global__ void hgt_attn_alpha(const float* __restrict__ q, const float* __restrict__ krel,
                               const int* __restrict__ src, const int* __restrict__ dst,
                               const float* __restrict__ prel,
                               float* __restrict__ alpha, unsigned* __restrict__ segmax, int E)
{
  long long t = (long long)blockIdx.x * blockDim.x + threadIdx.x;
  if (t >= (long long)E * NHEAD) return;
  int e = (int)(t >> 3), h = (int)(t & 7);
  int s = src[e], d = dst[e];
  const float4* qp = (const float4*)(q + (size_t)d * HID + h * DHEAD);
  const float4* kp = (const float4*)(krel + (size_t)s * HID + h * DHEAD);
  float acc = 0.f;
  #pragma unroll
  for (int i = 0; i < 8; ++i) {
    float4 a = qp[i], bb = kp[i];
    acc += a.x * bb.x + a.y * bb.y + a.z * bb.z + a.w * bb.w;
  }
  acc = acc * prel[h] * 0.17677669529663687f;   // 1/sqrt(32)
  alpha[t] = acc;
  atomicMax(segmax + (size_t)d * NHEAD + h, fkey(acc));
}

__global__ void hgt_attn_expsum(float* __restrict__ alpha, const int* __restrict__ dst,
                                const unsigned* __restrict__ segmax,
                                float* __restrict__ segsum, int E)
{
  long long t = (long long)blockIdx.x * blockDim.x + threadIdx.x;
  if (t >= (long long)E * NHEAD) return;
  int e = (int)(t >> 3), h = (int)(t & 7);
  int d = dst[e];
  float m = funkey(segmax[(size_t)d * NHEAD + h]);
  float ex = expf(alpha[t] - m);
  alpha[t] = ex;
  atomicAdd(segsum + (size_t)d * NHEAD + h, ex);
}

__global__ void hgt_attn_scatter(const float* __restrict__ alpha, const float* __restrict__ vrel,
                                 const int* __restrict__ src, const int* __restrict__ dst,
                                 const float* __restrict__ segsum,
                                 float* __restrict__ agg, int E)
{
  long long t = (long long)blockIdx.x * blockDim.x + threadIdx.x;
  if (t >= (long long)E * NHEAD) return;
  int e = (int)(t >> 3), h = (int)(t & 7);
  int s = src[e], d = dst[e];
  float w = alpha[t] / (segsum[(size_t)d * NHEAD + h] + 1e-16f);
  const float4* vp = (const float4*)(vrel + (size_t)s * HID + h * DHEAD);
  float* ap = agg + (size_t)d * HID + h * DHEAD;
  #pragma unroll
  for (int i = 0; i < 8; ++i) {
    float4 v = vp[i];
    atomicAdd(ap + 4 * i + 0, v.x * w);
    atomicAdd(ap + 4 * i + 1, v.y * w);
    atomicAdd(ap + 4 * i + 2, v.z * w);
    atomicAdd(ap + 4 * i + 3, v.w * w);
  }
}

// exact GELU, writes bf16 (input to the following A-projection GEMM)
__global__ void hgt_gelu_bf(const float* __restrict__ x, __bf16* __restrict__ y,
                            long long n) {
  long long i = (long long)blockIdx.x * blockDim.x + threadIdx.x;
  if (i < n) {
    float v = x[i];
    y[i] = f2bf(0.5f * v * (1.0f + erff(v * 0.70710678118654752f)));
  }
}

// x = a*o + (1-a)*x, also emits bf16 mirror of x for next layer's GEMMs
__global__ void hgt_combine(float* __restrict__ x, __bf16* __restrict__ xb,
                            const float* __restrict__ o,
                            const float* __restrict__ skip, long long n) {
  long long i = (long long)blockIdx.x * blockDim.x + threadIdx.x;
  if (i < n) {
    float a = 1.0f / (1.0f + expf(-skip[0]));
    float r = a * o[i] + (1.0f - a) * x[i];
    x[i] = r;
    xb[i] = f2bf(r);
  }
}

// ---------------------------------------------------------------------------
// Host orchestration
// ---------------------------------------------------------------------------
extern "C" void kernel_launch(void* const* d_in, const int* in_sizes, int n_in,
                              void* d_out, int out_size, void* d_ws, size_t ws_size,
                              hipStream_t stream)
{
  (void)in_sizes; (void)n_in; (void)out_size; (void)ws_size;
  const int NC = 50000, NH = 10000, E = 300000;

  const float* x_conn = (const float*)d_in[0];
  const float* x_host = (const float*)d_in[1];
  const int* src_h2c = (const int*)d_in[2];
  const int* dst_h2c = (const int*)d_in[3];
  const int* src_c2h = (const int*)d_in[4];
  const int* dst_c2h = (const int*)d_in[5];
  auto P = [&](int i) { return (const float*)d_in[i]; };

  // ---- workspace carve: f32 pool, then bf16 pool (all sizes 32B-aligned) ----
  float* wp = (float*)d_ws;
  auto alloc = [&](size_t ne) { float* r = wp; wp += ne; return r; };
  float* xc   = alloc((size_t)NC * HID);
  float* xh   = alloc((size_t)NH * HID);
  float* qc   = alloc((size_t)NC * HID);
  float* qh   = alloc((size_t)NH * HID);
  float* krel = alloc((size_t)NC * HID);   // also reused as conn A-proj output
  float* vrel = alloc((size_t)NC * HID);   // also reused as host A-proj output
  float* alph = alloc((size_t)E * NHEAD);
  unsigned* segmax = (unsigned*)alloc((size_t)NC * NHEAD);
  float* segsum = alloc((size_t)NC * NHEAD);
  float* aggc = alloc((size_t)NC * HID);
  float* aggh = alloc((size_t)NH * HID);

  __bf16* bp = (__bf16*)wp;
  auto balloc = [&](size_t ne) { __bf16* r = bp; bp += ne; return r; };
  __bf16* xc_bf  = balloc((size_t)NC * HID);
  __bf16* xh_bf  = balloc((size_t)NH * HID);
  __bf16* kc_bf  = balloc((size_t)NC * HID);
  __bf16* vc_bf  = balloc((size_t)NC * HID);
  __bf16* kh_bf  = balloc((size_t)NH * HID);
  __bf16* vh_bf  = balloc((size_t)NH * HID);
  __bf16* gc_bf  = balloc((size_t)NC * HID);
  __bf16* gh_bf  = balloc((size_t)NH * HID);
  __bf16* xinc_bf = balloc((size_t)NC * 64);
  __bf16* xinh_bf = balloc((size_t)NH * 32);
  __bf16* wt_in_c = balloc(64 * 256);
  __bf16* wt_in_h = balloc(32 * 256);
  __bf16* wt_proj[2][8];   // k_c,q_c,v_c,a_c,k_h,q_h,v_h,a_h
  __bf16* wt_rel[2][4];    // arel_h2c, mrel_h2c, arel_c2h, mrel_c2h
  for (int l = 0; l < 2; ++l) {
    for (int i = 0; i < 8; ++i) wt_proj[l][i] = balloc(256 * 256);
    for (int i = 0; i < 4; ++i) wt_rel[l][i] = balloc(NHEAD * 32 * 32);
  }
  __bf16* wt_out = balloc(256 * 8);

  dim3 blk(128);
  auto gemm = [&](const __bf16* A, int lda, long long sA,
                  const __bf16* Wt, long long sW, const float* bias,
                  float* Cf, __bf16* Cb, int ldc, long long sC,
                  int M, int N, int K, int act, int batch) {
    dim3 grid((unsigned)((M + 63) / 64), (unsigned)((N + 63) / 64), (unsigned)batch);
    hgt_gemm_wmma<<<grid, blk, 0, stream>>>(A, lda, sA, Wt, sW, bias,
                                            Cf, Cb, ldc, sC, M, N, K, act);
  };
  auto cvtW = [&](const float* W, __bf16* O, int K, int N, int batch) {
    dim3 grid((unsigned)((K * N + 255) / 256), 1, (unsigned)batch);
    hgt_cvt_wT<<<grid, 256, 0, stream>>>(W, O, K, N);
  };
  auto fillf = [&](float* ptr, float v, long long n) {
    hgt_fill_f32<<<(unsigned)((n + 255) / 256), 256, 0, stream>>>(ptr, v, n);
  };
  auto fillu = [&](unsigned* ptr, unsigned v, long long n) {
    hgt_fill_u32<<<(unsigned)((n + 255) / 256), 256, 0, stream>>>(ptr, v, n);
  };
  auto ew = [&](long long n) { return (unsigned)((n + 255) / 256); };

  // ---- one-time conversions: inputs and all weights (bf16, transposed) ----
  hgt_cvt_bf<<<ew((long long)NC * 64), 256, 0, stream>>>(x_conn, xinc_bf, (long long)NC * 64);
  hgt_cvt_bf<<<ew((long long)NH * 32), 256, 0, stream>>>(x_host, xinh_bf, (long long)NH * 32);
  cvtW(P(6), wt_in_c, 64, 256, 1);
  cvtW(P(8), wt_in_h, 32, 256, 1);
  for (int l = 0; l < 2; ++l) {
    const int pb = 10 + l * 24;
    const int widx[8] = {pb + 0, pb + 2, pb + 4, pb + 6, pb + 9, pb + 11, pb + 13, pb + 15};
    for (int i = 0; i < 8; ++i) cvtW(P(widx[i]), wt_proj[l][i], 256, 256, 1);
    cvtW(P(pb + 18), wt_rel[l][0], 32, 32, NHEAD);
    cvtW(P(pb + 19), wt_rel[l][1], 32, 32, NHEAD);
    cvtW(P(pb + 21), wt_rel[l][2], 32, 32, NHEAD);
    cvtW(P(pb + 22), wt_rel[l][3], 32, 32, NHEAD);
  }
  cvtW(P(58), wt_out, 256, 8, 1);

  // ---- input linears + ReLU (emit f32 + bf16 mirror) ----
  gemm(xinc_bf, 64, 0, wt_in_c, 0, P(7), xc, xc_bf, 256, 0, NC, 256, 64, 1, 1);
  gemm(xinh_bf, 32, 0, wt_in_h, 0, P(9), xh, xh_bf, 256, 0, NH, 256, 32, 1, 1);

  const long long EH = (long long)E * NHEAD;
  const unsigned egrid = (unsigned)((EH + 255) / 256);

  for (int l = 0; l < 2; ++l) {
    const int pb = 10 + l * 24;

    // K/V projections -> bf16 only (einsum inputs); Q -> f32 (attention dot)
    gemm(xc_bf, 256, 0, wt_proj[l][0], 0, P(pb + 1),  nullptr, kc_bf, 256, 0, NC, 256, 256, 0, 1);
    gemm(xc_bf, 256, 0, wt_proj[l][1], 0, P(pb + 3),  qc, nullptr,    256, 0, NC, 256, 256, 0, 1);
    gemm(xc_bf, 256, 0, wt_proj[l][2], 0, P(pb + 5),  nullptr, vc_bf, 256, 0, NC, 256, 256, 0, 1);
    gemm(xh_bf, 256, 0, wt_proj[l][4], 0, P(pb + 10), nullptr, kh_bf, 256, 0, NH, 256, 256, 0, 1);
    gemm(xh_bf, 256, 0, wt_proj[l][5], 0, P(pb + 12), qh, nullptr,    256, 0, NH, 256, 256, 0, 1);
    gemm(xh_bf, 256, 0, wt_proj[l][6], 0, P(pb + 14), nullptr, vh_bf, 256, 0, NH, 256, 256, 0, 1);

    fillf(aggc, 0.0f, (long long)NC * HID);
    fillf(aggh, 0.0f, (long long)NH * HID);

    auto relation = [&](const __bf16* kd_src, const __bf16* vd_src, const float* qd_dst,
                        const int* src, const int* dst, int n_src, int n_dst,
                        const __bf16* arelT, const __bf16* mrelT, const float* prel,
                        float* agg) {
      // per-head 32x32 transforms on source nodes (batched WMMA, grid.z = H)
      gemm(kd_src, 256, 32, arelT, 1024, nullptr, krel, nullptr, 256, 32, n_src, 32, 32, 0, NHEAD);
      gemm(vd_src, 256, 32, mrelT, 1024, nullptr, vrel, nullptr, 256, 32, n_src, 32, 32, 0, NHEAD);
      fillu(segmax, 0u, (long long)n_dst * NHEAD);
      fillf(segsum, 0.0f, (long long)n_dst * NHEAD);
      hgt_attn_alpha<<<egrid, 256, 0, stream>>>(qd_dst, krel, src, dst, prel, alph, segmax, E);
      hgt_attn_expsum<<<egrid, 256, 0, stream>>>(alph, dst, segmax, segsum, E);
      hgt_attn_scatter<<<egrid, 256, 0, stream>>>(alph, vrel, src, dst, segsum, agg, E);
    };

    relation(kh_bf, vh_bf, qc, src_h2c, dst_h2c, NH, NC,
             wt_rel[l][0], wt_rel[l][1], P(pb + 20), aggc);
    relation(kc_bf, vc_bf, qh, src_c2h, dst_c2h, NC, NH,
             wt_rel[l][2], wt_rel[l][3], P(pb + 23), aggh);

    // GELU -> A-projection -> gated skip (krel/vrel reused as f32 proj scratch)
    hgt_gelu_bf<<<ew((long long)NC * HID), 256, 0, stream>>>(aggc, gc_bf, (long long)NC * HID);
    gemm(gc_bf, 256, 0, wt_proj[l][3], 0, P(pb + 7), krel, nullptr, 256, 0, NC, 256, 256, 0, 1);
    hgt_combine<<<ew((long long)NC * HID), 256, 0, stream>>>(xc, xc_bf, krel, P(pb + 8),
                                                             (long long)NC * HID);

    hgt_gelu_bf<<<ew((long long)NH * HID), 256, 0, stream>>>(aggh, gh_bf, (long long)NH * HID);
    gemm(gh_bf, 256, 0, wt_proj[l][7], 0, P(pb + 16), vrel, nullptr, 256, 0, NH, 256, 256, 0, 1);
    hgt_combine<<<ew((long long)NH * HID), 256, 0, stream>>>(xh, xh_bf, vrel, P(pb + 17),
                                                             (long long)NH * HID);
  }

  // ---- output linear: [50000,256] @ [256,8] + b ----
  gemm(xc_bf, 256, 0, wt_out, 0, P(59), (float*)d_out, nullptr, 8, 0, NC, 8, 256, 0, 1);
}
